// QuantLinear_28973849379104
// MI455X (gfx1250) — compile-verified, matrix operand order checked by reading, not run on hardware
//
#include <hip/hip_runtime.h>

typedef __attribute__((ext_vector_type(16))) _Float16 v16h;
typedef __attribute__((ext_vector_type(8)))  _Float16 v8h;
typedef __attribute__((ext_vector_type(2)))  _Float16 v2h;
typedef __attribute__((ext_vector_type(8)))  float    v8f;

#define K_DIM     4096
#define N_DIM     11008
#define M_DIM     8192
#define GROUPSZ   128
#define BT_M      256
#define BT_N      128
#define BT_K      64
#define NTHREADS  512

// Fused 4-bit dequant + GEMM using v_wmma_f32_16x16x32_f16.
// Grid: (N/128, M/256), block: 512 threads (16 wave32 waves, 8(M) x 2(N)).
// K-order inside each group of 8 is permuted {0,4,1,5,2,6,3,7} consistently
// in BOTH LDS tiles so the packed-f16 nibble trick needs no re-shuffle.
__global__ __launch_bounds__(NTHREADS)
void gptq_gemm_wmma(const float* __restrict__ x,
                    const int*   __restrict__ qweight,
                    const int*   __restrict__ zeros,
                    const float* __restrict__ scales,
                    float* __restrict__ out)
{
    __shared__ __align__(16) _Float16 As[BT_M * BT_K];   // 32 KB, [m][k]
    __shared__ __align__(16) _Float16 Bs[BT_N * BT_K];   // 16 KB, [n][k]

    const int tid   = threadIdx.x;
    const int lane  = tid & 31;
    const int wid   = tid >> 5;
    const int m0    = blockIdx.y * BT_M;
    const int n0    = blockIdx.x * BT_N;

    const int waveM = (wid & 7) * 32;    // 8 waves along M
    const int waveN = (wid >> 3) * 64;   // 2 waves along N
    const int lan16 = lane & 15;
    const int hi    = lane >> 4;         // half-wave K-interleave selector

    // B-dequant mapping: thread owns one column, 2 of 8 qweight rows.
    const int bcol  = tid & 127;
    const int brow0 = (tid >> 7) * 2;
    const int ncol  = n0 + bcol;

    v8f acc[2][4] = {};                  // 2(M) x 4(N) 16x16 f32 tiles

    for (int k0 = 0; k0 < K_DIM; k0 += BT_K) {
        __syncthreads();

        // ---- A tile: 256x64 f32 -> f16, permuted pair order ----
        #pragma unroll
        for (int i = 0; i < 4; ++i) {
            const int idx = tid + i * NTHREADS;     // 0..2047 8-float chunks
            const int row = idx >> 3;               // 0..255
            const int c8  = (idx & 7) * 8;          // 0..56
            const float* xp = x + (size_t)(m0 + row) * K_DIM + k0 + c8;
            const float4 f0 = *(const float4*)(xp);
            const float4 f1 = *(const float4*)(xp + 4);
            v8h h;                                   // k order {0,4,1,5,2,6,3,7}
            h[0] = (_Float16)f0.x; h[1] = (_Float16)f1.x;
            h[2] = (_Float16)f0.y; h[3] = (_Float16)f1.y;
            h[4] = (_Float16)f0.z; h[5] = (_Float16)f1.z;
            h[6] = (_Float16)f0.w; h[7] = (_Float16)f1.w;
            *(v8h*)(&As[row * BT_K + c8]) = h;
        }

        // ---- B tile: packed-f16 nibble dequant, 2 words/thread ----
        const int g  = k0 / GROUPSZ;
        const int zw = zeros[g * (N_DIM / 8) + (ncol >> 3)];
        const unsigned zn = (unsigned)(zw >> ((ncol & 7) * 4)) & 0xFu;
        // zero as f16 pair (1024+z, 1024+z): exact Sterbenz subtraction below
        const unsigned zbits = 0x64006400u | (zn << 16) | zn;
        const v2h zh = __builtin_bit_cast(v2h, zbits);
        const _Float16 sh = (_Float16)scales[g * N_DIM + ncol];
        const v2h sv = { sh, sh };
        #pragma unroll
        for (int j = 0; j < 2; ++j) {
            const int qr = brow0 + j;                           // 0..7
            const unsigned q =
                (unsigned)qweight[(size_t)(k0 / 8 + qr) * N_DIM + ncol];
            // pairs (n0,n4) (n1,n5) (n2,n6) (n3,n7), each -> f16 1024+n
            v2h p0 = __builtin_bit_cast(v2h, (q         & 0x000F000Fu) | 0x64006400u);
            v2h p1 = __builtin_bit_cast(v2h, ((q >> 4)  & 0x000F000Fu) | 0x64006400u);
            v2h p2 = __builtin_bit_cast(v2h, ((q >> 8)  & 0x000F000Fu) | 0x64006400u);
            v2h p3 = __builtin_bit_cast(v2h, ((q >> 12) & 0x000F000Fu) | 0x64006400u);
            p0 = (p0 - zh) * sv;                    // v_pk_add_f16(neg)+v_pk_mul_f16
            p1 = (p1 - zh) * sv;
            p2 = (p2 - zh) * sv;
            p3 = (p3 - zh) * sv;
            v8h h;
            h[0] = p0[0]; h[1] = p0[1];
            h[2] = p1[0]; h[3] = p1[1];
            h[4] = p2[0]; h[5] = p2[1];
            h[6] = p3[0]; h[7] = p3[1];
            *(v8h*)(&Bs[bcol * BT_K + qr * 8]) = h;             // 16B store
        }

        // Prefetch next K-step tiles into L2 while computing this one.
        if (k0 + BT_K < K_DIM) {
            __builtin_prefetch(x + (size_t)(m0 + (tid >> 1)) * K_DIM + k0 + BT_K, 0, 1);
            __builtin_prefetch(qweight + (size_t)((k0 + BT_K) / 8 + brow0) * N_DIM + ncol, 0, 1);
        }

        __syncthreads();

        // ---- compute: 2 K=32 WMMA steps, 2x4 tiles each ----
        #pragma unroll
        for (int kk = 0; kk < BT_K; kk += 32) {
            v16h a[2], b[4];
            #pragma unroll
            for (int mt = 0; mt < 2; ++mt) {
                const _Float16* p =
                    &As[(waveM + mt * 16 + lan16) * BT_K + kk + hi * 8];
                v8h lo = *(const v8h*)p;            // K 0-7  / 8-15  (permuted)
                v8h hh = *(const v8h*)(p + 16);     // K 16-23 / 24-31 (permuted)
                a[mt] = __builtin_shufflevector(lo, hh,
                        0,1,2,3,4,5,6,7,8,9,10,11,12,13,14,15);
            }
            #pragma unroll
            for (int nt = 0; nt < 4; ++nt) {
                const _Float16* p =
                    &Bs[(waveN + nt * 16 + lan16) * BT_K + kk + hi * 8];
                v8h lo = *(const v8h*)p;
                v8h hh = *(const v8h*)(p + 16);
                b[nt] = __builtin_shufflevector(lo, hh,
                        0,1,2,3,4,5,6,7,8,9,10,11,12,13,14,15);
            }
            #pragma unroll
            for (int mt = 0; mt < 2; ++mt) {
                #pragma unroll
                for (int nt = 0; nt < 4; ++nt) {
                    acc[mt][nt] = __builtin_amdgcn_wmma_f32_16x16x32_f16(
                        false, a[mt], false, b[nt],
                        (short)0, acc[mt][nt], false, false);
                }
            }
        }
    }

    // ---- store: C/D layout: VGPR r -> M = r + 8*hi, N = lane&15 ----
    #pragma unroll
    for (int mt = 0; mt < 2; ++mt) {
        #pragma unroll
        for (int nt = 0; nt < 4; ++nt) {
            const int col = n0 + waveN + nt * 16 + lan16;
            #pragma unroll
            for (int r = 0; r < 8; ++r) {
                const int row = m0 + waveM + mt * 16 + hi * 8 + r;
                out[(size_t)row * N_DIM + col] = acc[mt][nt][r];
            }
        }
    }
}

extern "C" void kernel_launch(void* const* d_in, const int* in_sizes, int n_in,
                              void* d_out, int out_size, void* d_ws, size_t ws_size,
                              hipStream_t stream) {
    const float* x       = (const float*)d_in[0];
    const int*   qweight = (const int*)d_in[1];
    const int*   zeros   = (const int*)d_in[2];
    const float* scales  = (const float*)d_in[3];
    float*       out     = (float*)d_out;

    dim3 grid(N_DIM / BT_N, M_DIM / BT_M);   // 86 x 32 workgroups
    gptq_gemm_wmma<<<grid, NTHREADS, 0, stream>>>(x, qweight, zeros, scales, out);
}
